// TransformerBlock_41910290874538
// MI455X (gfx1250) — matrix operand order repaired and checked by reading, not compiled, not run on hardware
//
#include <hip/hip_runtime.h>
#include <hip/hip_bf16.h>

#define SEQ    2048
#define BATCH  2
#define DMODEL 1024
#define NHEADS 16
#define HDIM   64
#define DFF    4096

typedef __attribute__((ext_vector_type(16))) __bf16 bf16x16;
typedef __attribute__((ext_vector_type(8)))  float  f32x8;
typedef __attribute__((ext_vector_type(4)))  __bf16 bf16x4;
typedef __attribute__((ext_vector_type(4)))  unsigned int tdm_u32x4;
typedef __attribute__((ext_vector_type(4)))  int          tdm_i32x4;
typedef __attribute__((ext_vector_type(8)))  int          tdm_i32x8;

// ROCm 7.2 (clang-22): 5-arg tensor_load_to_lds builtin.
// amdgpu-toolchain (therock-10.0 headers present): 6-arg form.
#if defined(__has_include)
#if __has_include(<hip/amd_detail/amd_gfx1250_TDM.h>)
#define TDM_ARITY6 1
#endif
#endif
#ifndef TDM_ARITY6
#define TDM_ARITY6 0
#endif

union FragAB {
  uint4   u[2];
  bf16x16 v;
};

static __device__ __forceinline__ f32x8 wmma_bf16(bf16x16 a, bf16x16 b, f32x8 c) {
  // D = A(16x32 bf16) * B(32x16 bf16) + C(16x16 f32)
  return __builtin_amdgcn_wmma_f32_16x16x32_bf16(false, a, false, b, (short)0, c,
                                                 false, false);
}

// ---------------------------------------------------------------------------
// DPP16 row-rotate reductions: reduce across the 16 lanes of each half-wave
// (rows of 16) without touching the LDS pipe (vs __shfl_xor -> ds_bpermute).
// ROW_ROR:n control = 0x120 | n; rotate-reduce by 1,2,4,8 covers the row.
// ---------------------------------------------------------------------------
template <int CTRL>
static __device__ __forceinline__ float dpp_mov_f32(float x) {
  return __int_as_float(
      __builtin_amdgcn_update_dpp(0, __float_as_int(x), CTRL, 0xf, 0xf, true));
}
static __device__ __forceinline__ float row16_max(float x) {
  x = fmaxf(x, dpp_mov_f32<0x121>(x));  // row_ror:1
  x = fmaxf(x, dpp_mov_f32<0x122>(x));  // row_ror:2
  x = fmaxf(x, dpp_mov_f32<0x124>(x));  // row_ror:4
  x = fmaxf(x, dpp_mov_f32<0x128>(x));  // row_ror:8
  return x;
}
static __device__ __forceinline__ float row16_sum(float x) {
  x += dpp_mov_f32<0x121>(x);
  x += dpp_mov_f32<0x122>(x);
  x += dpp_mov_f32<0x124>(x);
  x += dpp_mov_f32<0x128>(x);
  return x;
}

// ---------------------------------------------------------------------------
// TDM: DMA a 128-row x 32-element bf16 tile (row stride = stride elems) from
// global into LDS at lds_off.  Pad: 64B data rows + 16B pad -> 80B LDS row
// stride (40 bf16) so ds_load_b128 fragment fetches are bank-conflict free.
// D# layout per CDNA5 ISA ch.10.8 (group0 128b, group1 256b; 2D => groups 2/3
// disabled / zero).
// ---------------------------------------------------------------------------
static __device__ __forceinline__ void tdm_load_slab(unsigned int lds_off,
                                                     const __bf16* gptr,
                                                     int stride_elems) {
  const unsigned long long ga = (unsigned long long)(uintptr_t)gptr;
  const int rows = 128, cols = 32;
  tdm_u32x4 g0;
  g0[0] = 1u;                                        // count=1, user mode
  g0[1] = lds_off;                                   // lds_addr (bytes)
  g0[2] = (unsigned int)(ga & 0xFFFFFFFFu);          // global_addr[31:0]
  g0[3] = (unsigned int)((ga >> 32) & 0x01FFFFFFu)   // global_addr[56:32]
          | (2u << 30);                              // type=2 (image)
  tdm_i32x8 g1;
  g1[0] = (1 << 16)      // data_size = 1 -> 2 bytes
        | (1 << 20)      // pad_enable
        | (3 << 22)      // pad_interval: 1<<3 8-byte units = 64B of data
        | (3 << 25);     // pad_amount: 4 DWORDs = 16B
  g1[1] = (cols & 0xFFFF) << 16;                     // tensor_dim0[15:0]
  g1[2] = ((rows & 0xFFFF) << 16);                   // dim0 hi=0 | tensor_dim1 lo
  g1[3] = ((cols & 0xFFFF) << 16);                   // dim1 hi=0 | tile_dim0
  g1[4] = (rows & 0xFFFF);                           // tile_dim1 | tile_dim2=0
  g1[5] = stride_elems;                              // tensor_dim0_stride[31:0]
  g1[6] = 0;                                         // stride hi | dim1_stride lo
  g1[7] = 0;
  tdm_i32x4 z4 = {0, 0, 0, 0};
#if TDM_ARITY6
  tdm_i32x8 z8 = {0, 0, 0, 0, 0, 0, 0, 0};
  __builtin_amdgcn_tensor_load_to_lds(g0, g1, z4, z4, z8, 0);
#else
  __builtin_amdgcn_tensor_load_to_lds(g0, g1, z4, z4, 0);
#endif
}

// ---------------------------------------------------------------------------
// Weight conversion: fp32 (bt,R,C) -> bf16 transposed (bt,C,R), times scale.
// (scale = 1/sqrt(HDIM) for Wq folds the softmax scaling into Q.)
// ---------------------------------------------------------------------------
__global__ void k_transpose_f32_bf16(const float* __restrict__ in,
                                     __bf16* __restrict__ out,
                                     int R, int C, int total, float scale) {
  int i = blockIdx.x * blockDim.x + threadIdx.x;
  if (i >= total) return;
  int c  = i % C;
  int t  = i / C;
  int r  = t % R;
  int bt = t / R;
  out[(size_t)bt * R * C + (size_t)c * R + r] = (__bf16)(in[i] * scale);
}

// ---------------------------------------------------------------------------
// RMSNorm: fp32 row (1024) -> bf16 row.  One 256-thread block per row.
// ---------------------------------------------------------------------------
__global__ void k_rmsnorm_bf16(const float* __restrict__ x,
                               const float* __restrict__ g,
                               __bf16* __restrict__ out) {
  int row = blockIdx.x;
  int tid = threadIdx.x;  // 0..255, 4 floats each
  const float4 v = ((const float4*)(x + (size_t)row * DMODEL))[tid];
  float ss = v.x * v.x + v.y * v.y + v.z * v.z + v.w * v.w;
  ss = row16_sum(ss);                       // within each 16-lane row (DPP)
  ss += __shfl_xor(ss, 16, 32);             // cross-half once
  __shared__ float red[8];
  if ((tid & 31) == 0) red[tid >> 5] = ss;
  __syncthreads();
  float tot = 0.f;
#pragma unroll
  for (int i = 0; i < 8; ++i) tot += red[i];
  float inv = rsqrtf(tot * (1.0f / DMODEL) + 1e-6f);
  const float4 gv = ((const float4*)g)[tid];
  bf16x4 o;
  o[0] = (__bf16)(v.x * inv * gv.x);
  o[1] = (__bf16)(v.y * inv * gv.y);
  o[2] = (__bf16)(v.z * inv * gv.z);
  o[3] = (__bf16)(v.w * inv * gv.w);
  *(bf16x4*)(out + (size_t)row * DMODEL + (size_t)tid * 4) = o;
}

// ---------------------------------------------------------------------------
// GEMM: C(MxN) = A(MxK, bf16 row-major) * Bt(NxK, bf16 row-major == B^T)
// Workgroup = 256 threads = 8 waves, 128x128 C-tile; each wave owns 64x32.
// K-slabs (128x32) of A and B are TDM-DMA'd into double-buffered LDS (80B row
// stride), synced with s_wait_tensorcnt + workgroup barrier; fragments come
// from LDS via ds_load_b128; math is v_wmma_f32_16x16x32_bf16.
// MODE 0: bf16 row-major out             MODE 1: Q/K scatter (b,h,s,d)
// MODE 2: V scatter transposed (b,h,d,s) MODE 3: f32 out = Res + acc
// MODE 4: bf16 out = silu(acc)
// ---------------------------------------------------------------------------
#define LDS_ROW 40  // bf16 elems per LDS slab row (32 data + 4-dword TDM pad)

template <int MODE>
__global__ __launch_bounds__(256) void k_gemm(const __bf16* __restrict__ A, int lda,
                                              const __bf16* __restrict__ Bt, int ldb,
                                              void* __restrict__ Cout,
                                              const float* __restrict__ Res,
                                              int K, int ldc) {
  const int tid = threadIdx.x;
  const int lane = tid & 31, wid = tid >> 5;
  const int lm = lane & 15, half = lane >> 4;
  const int waveM = wid >> 2, waveN = wid & 3;        // 2 x 4 wave grid
  const int tN = blockIdx.x * 128, tM = blockIdx.y * 128;

  __shared__ __align__(16) __bf16 sA[2][128 * LDS_ROW];
  __shared__ __align__(16) __bf16 sB[2][128 * LDS_ROW];

  f32x8 acc[4][2] = {};

  if (wid == 0) {
    tdm_load_slab((unsigned int)(uintptr_t)&sA[0][0], A + (size_t)tM * lda, lda);
    tdm_load_slab((unsigned int)(uintptr_t)&sB[0][0], Bt + (size_t)tN * ldb, ldb);
    __builtin_amdgcn_s_wait_tensorcnt(0);
  }
  __syncthreads();

  for (int k0 = 0; k0 < K; k0 += 32) {
    const int cur = (k0 >> 5) & 1;
    const bool more = (k0 + 32) < K;
    if (wid == 0 && more) {  // prefetch next slabs into the other buffer
      tdm_load_slab((unsigned int)(uintptr_t)&sA[cur ^ 1][0],
                    A + (size_t)tM * lda + k0 + 32, lda);
      tdm_load_slab((unsigned int)(uintptr_t)&sB[cur ^ 1][0],
                    Bt + (size_t)tN * ldb + k0 + 32, ldb);
    }

    FragAB a[4];
#pragma unroll
    for (int i = 0; i < 4; ++i) {
      const __bf16* p = &sA[cur][(size_t)(64 * waveM + 16 * i + lm) * LDS_ROW];
      a[i].u[0] = *(const uint4*)(p + 8 * half);       // A: k = j + 8*half
      a[i].u[1] = *(const uint4*)(p + 16 + 8 * half);  //    k = j + 8 + 8*half
    }
    FragAB b[2];
#pragma unroll
    for (int j = 0; j < 2; ++j) {
      const __bf16* p =
          &sB[cur][(size_t)(32 * waveN + 16 * j + lm) * LDS_ROW] + 8 * half;
      b[j].u[0] = *(const uint4*)(p);                  // B: k = j + 8*half
      b[j].u[1] = *(const uint4*)(p + 16);
    }
#pragma unroll
    for (int i = 0; i < 4; ++i)
#pragma unroll
      for (int j = 0; j < 2; ++j) acc[i][j] = wmma_bf16(a[i].v, b[j].v, acc[i][j]);

    if (wid == 0 && more) __builtin_amdgcn_s_wait_tensorcnt(0);
    __syncthreads();
  }

#pragma unroll
  for (int i = 0; i < 4; ++i)
#pragma unroll
    for (int j = 0; j < 2; ++j)
#pragma unroll
      for (int r = 0; r < 8; ++r) {
        const int m = tM + 64 * waveM + 16 * i + r + 8 * half;  // C: m = r+8*half
        const int n = tN + 32 * waveN + 16 * j + lm;            //    n = lane%16
        const float v = acc[i][j][r];
        if constexpr (MODE == 0) {
          ((__bf16*)Cout)[(size_t)m * ldc + n] = (__bf16)v;
        } else if constexpr (MODE == 1) {  // rows (s*BATCH+b), cols h*64+k
          int s = m >> 1, bb = m & 1, hh = n >> 6, kk = n & 63;
          ((__bf16*)Cout)[(((size_t)(bb * NHEADS + hh)) * SEQ + s) * HDIM + kk] =
              (__bf16)v;
        } else if constexpr (MODE == 2) {  // V^T: (b,h,d,s)
          int s = m >> 1, bb = m & 1, hh = n >> 6, kk = n & 63;
          ((__bf16*)Cout)[(((size_t)(bb * NHEADS + hh)) * HDIM + kk) * SEQ + s] =
              (__bf16)v;
        } else if constexpr (MODE == 3) {
          ((float*)Cout)[(size_t)m * ldc + n] = Res[(size_t)m * ldc + n] + v;
        } else {  // SiLU
          float sv = v / (1.0f + __expf(-v));
          ((__bf16*)Cout)[(size_t)m * ldc + n] = (__bf16)sv;
        }
      }
}

// ---------------------------------------------------------------------------
// Flash attention, one wave per (b*H+h, 32-query block).
// One 32-key block step; DIAG=true only for the diagonal block so the
// steady-state loop carries no causal-mask VALU at all.  Score scale is
// pre-folded into Wq.  Row reductions use DPP row_ror (no LDS traffic).
// ---------------------------------------------------------------------------
template <bool DIAG>
static __device__ __forceinline__ void flash_block(
    int kb, int q0, int lm, int half, const __bf16* __restrict__ Kp,
    const __bf16* __restrict__ Vp, FragAB (&qa)[2][2], f32x8 (&O)[2][4],
    float (&mrow)[2][8], float (&lrow)[2][8], __bf16 (*Pl)[32]) {
  f32x8 S[2][2] = {};
  // scores = Q . K^T  (B-operand column n = key row of K; k = d contiguous)
#pragma unroll
  for (int nt = 0; nt < 2; ++nt) {
    FragAB kf[2];
#pragma unroll
    for (int ks = 0; ks < 2; ++ks) {
      const __bf16* p = Kp + (size_t)(kb + 16 * nt + lm) * HDIM + 32 * ks + 8 * half;
      kf[ks].u[0] = *(const uint4*)(p);
      kf[ks].u[1] = *(const uint4*)(p + 16);
    }
#pragma unroll
    for (int mt = 0; mt < 2; ++mt)
#pragma unroll
      for (int ks = 0; ks < 2; ++ks)
        S[mt][nt] = wmma_bf16(qa[mt][ks].v, kf[ks].v, S[mt][nt]);
  }

#pragma unroll
  for (int mt = 0; mt < 2; ++mt) {
    if constexpr (DIAG) {
#pragma unroll
      for (int nt = 0; nt < 2; ++nt)
#pragma unroll
        for (int r = 0; r < 8; ++r) {
          int qi = q0 + 16 * mt + r + 8 * half;
          int ki = kb + 16 * nt + lm;
          if (ki > qi) S[mt][nt][r] = -1e30f;
        }
    }
    // row max across the 16 lanes of each half (DPP row_ror reduce)
    float rm[8];
#pragma unroll
    for (int r = 0; r < 8; ++r)
      rm[r] = row16_max(fmaxf(S[mt][0][r], S[mt][1][r]));

    float rs[8];
#pragma unroll
    for (int r = 0; r < 8; ++r) {
      float mn = fmaxf(mrow[mt][r], rm[r]);
      float al = __expf(mrow[mt][r] - mn);
      mrow[mt][r] = mn;
      lrow[mt][r] *= al;
#pragma unroll
      for (int nd = 0; nd < 4; ++nd) O[mt][nd][r] *= al;
      float p0 = __expf(S[mt][0][r] - mn);
      float p1 = __expf(S[mt][1][r] - mn);
      S[mt][0][r] = p0;
      S[mt][1][r] = p1;
      rs[r] = p0 + p1;
    }
#pragma unroll
    for (int r = 0; r < 8; ++r) lrow[mt][r] += row16_sum(rs[r]);

    // stage P (bf16) to LDS in row-major [query][key]
#pragma unroll
    for (int nt = 0; nt < 2; ++nt)
#pragma unroll
      for (int r = 0; r < 8; ++r)
        Pl[16 * mt + r + 8 * half][16 * nt + lm] = (__bf16)S[mt][nt][r];
  }
  __syncthreads();  // single-wave WG: barrier NOP, orders ds traffic

  // reload P in A-operand layout
  FragAB pa[2];
#pragma unroll
  for (int mt = 0; mt < 2; ++mt) {
    const __bf16* p = &Pl[16 * mt + lm][0];
    pa[mt].u[0] = *(const uint4*)(p + 8 * half);
    pa[mt].u[1] = *(const uint4*)(p + 16 + 8 * half);
  }
  __syncthreads();

  // O += P . V   (B-operand column n = d-row of V^T; k = key contiguous)
#pragma unroll
  for (int nd = 0; nd < 4; ++nd) {
    FragAB vb;
    const __bf16* p = Vp + (size_t)(16 * nd + lm) * SEQ + kb + 8 * half;
    vb.u[0] = *(const uint4*)(p);
    vb.u[1] = *(const uint4*)(p + 16);
#pragma unroll
    for (int mt = 0; mt < 2; ++mt) O[mt][nd] = wmma_bf16(pa[mt].v, vb.v, O[mt][nd]);
  }
}

// Q (b,h,s,d) bf16 (pre-scaled), K (b,h,s,d) bf16, V^T (b,h,d,s) bf16
//   -> ctx (s,b,h*64+d) bf16
__global__ __launch_bounds__(32) void k_flash(const __bf16* __restrict__ Q,
                                              const __bf16* __restrict__ Km,
                                              const __bf16* __restrict__ Vt,
                                              __bf16* __restrict__ ctx) {
  const int lane = threadIdx.x;
  const int lm = lane & 15, half = lane >> 4;
  const int qb = blockIdx.x;
  const int bh = blockIdx.y;
  const int bb = bh >> 4, hh = bh & 15;
  const __bf16* Qp = Q + (size_t)bh * SEQ * HDIM;
  const __bf16* Kp = Km + (size_t)bh * SEQ * HDIM;
  const __bf16* Vp = Vt + (size_t)bh * HDIM * SEQ;
  const int q0 = qb * 32;

  // Q fragments stay resident: 2 m-tiles x 2 k-steps (d = 0..63)
  FragAB qa[2][2];
#pragma unroll
  for (int mt = 0; mt < 2; ++mt)
#pragma unroll
    for (int ks = 0; ks < 2; ++ks) {
      const __bf16* p = Qp + (size_t)(q0 + 16 * mt + lm) * HDIM + 32 * ks;
      qa[mt][ks].u[0] = *(const uint4*)(p + 8 * half);
      qa[mt][ks].u[1] = *(const uint4*)(p + 16 + 8 * half);
    }

  f32x8 O[2][4] = {};
  float mrow[2][8], lrow[2][8];
#pragma unroll
  for (int mt = 0; mt < 2; ++mt)
#pragma unroll
    for (int r = 0; r < 8; ++r) { mrow[mt][r] = -1e30f; lrow[mt][r] = 0.f; }

  __shared__ __align__(16) __bf16 Pl[32][32];  // C-layout -> A-layout staging

  // full (unmasked) key blocks
  for (int jb = 0; jb < qb; ++jb) {
    const int kb = jb * 32;
    // prefetch next K/V tiles (global_prefetch_b8)
    __builtin_prefetch(Kp + (size_t)(kb + 32 + lane) * HDIM, 0, 1);
    __builtin_prefetch(Vp + (size_t)(2 * lane) * SEQ + kb + 32, 0, 1);
    flash_block<false>(kb, q0, lm, half, Kp, Vp, qa, O, mrow, lrow, Pl);
  }
  // diagonal (causally masked) block
  flash_block<true>(qb * 32, q0, lm, half, Kp, Vp, qa, O, mrow, lrow, Pl);

  // normalize and scatter to ctx (s, b, h*64+d)
#pragma unroll
  for (int mt = 0; mt < 2; ++mt)
#pragma unroll
    for (int nd = 0; nd < 4; ++nd)
#pragma unroll
      for (int r = 0; r < 8; ++r) {
        int qi = q0 + 16 * mt + r + 8 * half;
        float val = O[mt][nd][r] / lrow[mt][r];
        int d = hh * HDIM + 16 * nd + lm;
        ctx[((size_t)qi * BATCH + bb) * DMODEL + d] = (__bf16)val;
      }
}

// ---------------------------------------------------------------------------
extern "C" void kernel_launch(void* const* d_in, const int* in_sizes, int n_in,
                              void* d_out, int out_size, void* d_ws, size_t ws_size,
                              hipStream_t stream) {
  (void)in_sizes; (void)n_in; (void)out_size; (void)ws_size;
  const float* x  = (const float*)d_in[0];
  const float* g1 = (const float*)d_in[1];
  const float* g2 = (const float*)d_in[2];
  const float* Wq = (const float*)d_in[3];
  const float* Wk = (const float*)d_in[4];
  const float* Wv = (const float*)d_in[5];
  const float* Wo = (const float*)d_in[6];
  const float* W1 = (const float*)d_in[7];
  const float* W2 = (const float*)d_in[8];
  float* out = (float*)d_out;

  char* ws = (char*)d_ws;
  size_t off = 0;
  auto alloc = [&](size_t bytes) {
    void* p = ws + off;
    off += (bytes + 255) & ~(size_t)255;
    return p;
  };
  const size_t R = (size_t)SEQ * BATCH;  // 4096 rows

  __bf16* hA  = (__bf16*)alloc(R * DMODEL * 2);           // rmsnorm1(x)
  __bf16* WqT = (__bf16*)alloc((size_t)DMODEL * DMODEL * 2);
  __bf16* WkT = (__bf16*)alloc((size_t)DMODEL * DMODEL * 2);
  __bf16* WvT = (__bf16*)alloc((size_t)DMODEL * DMODEL * 2);
  __bf16* WoT = (__bf16*)alloc((size_t)DMODEL * DMODEL * 2);
  __bf16* W1T = (__bf16*)alloc((size_t)DMODEL * DFF * 2);
  __bf16* W2T = (__bf16*)alloc((size_t)DFF * DMODEL * 2);
  __bf16* Qb  = (__bf16*)alloc(R * DMODEL * 2);           // (b,h,s,d), pre-scaled
  __bf16* Kb  = (__bf16*)alloc(R * DMODEL * 2);           // (b,h,s,d)
  __bf16* Vb  = (__bf16*)alloc(R * DMODEL * 2);           // (b,h,d,s)
  __bf16* ctx = (__bf16*)alloc(R * DMODEL * 2);           // (s,b,D)
  float*  x1  = (float*)alloc(R * DMODEL * 4);            // x + ctx@Wo
  __bf16* h2  = (__bf16*)alloc(R * DMODEL * 2);           // rmsnorm2(x1)
  __bf16* y   = (__bf16*)alloc(R * DFF * 2);              // silu(h2@W1)

  // ---- weights -> bf16, transposed (B^T = N x K, k contiguous) ----
  // Wq also absorbs the 1/sqrt(HDIM) attention scale.
  {
    const float qscale = 0.125f;  // HDIM^-0.5
    int tot = NHEADS * DMODEL * HDIM;  // per-head (D,K) -> (K,D)
    k_transpose_f32_bf16<<<(tot + 255) / 256, 256, 0, stream>>>(Wq, WqT, DMODEL, HDIM, tot, qscale);
    k_transpose_f32_bf16<<<(tot + 255) / 256, 256, 0, stream>>>(Wk, WkT, DMODEL, HDIM, tot, 1.0f);
    k_transpose_f32_bf16<<<(tot + 255) / 256, 256, 0, stream>>>(Wv, WvT, DMODEL, HDIM, tot, 1.0f);
    tot = DMODEL * DMODEL;
    k_transpose_f32_bf16<<<(tot + 255) / 256, 256, 0, stream>>>(Wo, WoT, DMODEL, DMODEL, tot, 1.0f);
    tot = DMODEL * DFF;
    k_transpose_f32_bf16<<<(tot + 255) / 256, 256, 0, stream>>>(W1, W1T, DMODEL, DFF, tot, 1.0f);
    k_transpose_f32_bf16<<<(tot + 255) / 256, 256, 0, stream>>>(W2, W2T, DFF, DMODEL, tot, 1.0f);
  }

  // ---- attention ----
  k_rmsnorm_bf16<<<(int)R, 256, 0, stream>>>(x, g1, hA);
  dim3 gQKV(DMODEL / 128, (unsigned)(R / 128));  // (8, 32)
  k_gemm<1><<<gQKV, 256, 0, stream>>>(hA, DMODEL, WqT, DMODEL, Qb, nullptr, DMODEL, DMODEL);
  k_gemm<1><<<gQKV, 256, 0, stream>>>(hA, DMODEL, WkT, DMODEL, Kb, nullptr, DMODEL, DMODEL);
  k_gemm<2><<<gQKV, 256, 0, stream>>>(hA, DMODEL, WvT, DMODEL, Vb, nullptr, DMODEL, DMODEL);
  k_flash<<<dim3(SEQ / 32, BATCH * NHEADS), 32, 0, stream>>>(Qb, Kb, Vb, ctx);
  k_gemm<3><<<gQKV, 256, 0, stream>>>(ctx, DMODEL, WoT, DMODEL, x1, x, DMODEL, DMODEL);

  // ---- FFN ----
  k_rmsnorm_bf16<<<(int)R, 256, 0, stream>>>(x1, g2, h2);
  k_gemm<4><<<dim3(DFF / 128, (unsigned)(R / 128)), 256, 0, stream>>>(
      h2, DMODEL, W1T, DMODEL, y, nullptr, DMODEL, DFF);
  k_gemm<3><<<gQKV, 256, 0, stream>>>(y, DFF, W2T, DFF, out, x1, DFF, DMODEL);
}
